// LSTMModel_944892805211
// MI455X (gfx1250) — matrix operand order repaired and checked by reading, not compiled, not run on hardware
//
#include <hip/hip_runtime.h>

// ---------------------------------------------------------------------------
// Persistent, software-pipelined 3-layer LSTM for gfx1250 (MI455X).
//  - 48 workgroups x 256 threads: WG (16*l + bt) runs layer l on batch rows
//    [16*bt, 16*bt+16). Layers pipeline over t via agent-scope flags in ws.
//  - Recurrent + input GEMMs via v_wmma_f32_16x16x32_bf16; weight (B) fragments
//    preloaded to VGPRs once; h kept in LDS (bf16), c in LDS (f32).
//  - One shared [B,T,H] bf16 buffer (32 MB, L2-resident) reused in place by
//    the whole layer chain; head kernel reduces it to [B,T,1] f32.
// ---------------------------------------------------------------------------

typedef __bf16 bf16;
typedef __attribute__((ext_vector_type(16))) __bf16 v16bf;
typedef __attribute__((ext_vector_type(8)))  __bf16 v8bf;
typedef __attribute__((ext_vector_type(8)))  float  v8f;

#define B_   256
#define T_   1024
#define D_   18
#define H_   64
#define G_   256           // 4*H (gate columns, order i,f,g,o)
#define HP   72            // padded LDS row stride for h (bf16)
#define GP   260           // padded LDS row stride for gates (f32)
#define HC   68            // padded LDS row stride for c (f32)

#define WMMA_BF16(Afrag, Bfrag, Cfrag) \
  __builtin_amdgcn_wmma_f32_16x16x32_bf16(false, (Afrag), false, (Bfrag), (short)0, (Cfrag), false, false)

__device__ __forceinline__ float fsig(float x) {
  return 1.0f / (1.0f + __expf(-x));
}
__device__ __forceinline__ float ftanh(float x) {
  const float e = __expf(-2.0f * fabsf(x));
  const float r = (1.0f - e) / (1.0f + e);
  return copysignf(r, x);
}

union AU { v16bf v; v8bf h8[2]; };

__global__ __launch_bounds__(256)
void lstm3_fused(const float* __restrict__ x,
                 const float* __restrict__ W_ih0, const float* __restrict__ W_hh0,
                 const float* __restrict__ b_ih0, const float* __restrict__ b_hh0,
                 const float* __restrict__ W_ih12, const float* __restrict__ W_hh12,
                 const float* __restrict__ b_ih12, const float* __restrict__ b_hh12,
                 bf16* buf, unsigned* flags)
{
  const int tid  = threadIdx.x;
  const int wave = tid >> 5;
  const int lane = tid & 31;
  const int row  = lane & 15;     // M row (A) or N col (B/C) within tile
  const int hi   = lane >> 4;     // half-wave selector for WMMA layouts
  const int layer = blockIdx.x >> 4;
  const int btile = blockIdx.x & 15;

  const float *Wih, *Whh, *bih, *bhh;
  int Din;
  if (layer == 0) { Wih = W_ih0; Whh = W_hh0; bih = b_ih0; bhh = b_hh0; Din = D_; }
  else {
    const int l = layer - 1;
    Wih = W_ih12 + (size_t)l * G_ * H_;  Whh = W_hh12 + (size_t)l * G_ * H_;
    bih = b_ih12 + l * G_;               bhh = b_hh12 + l * G_;
    Din = H_;
  }

  __shared__ bf16  hlds[16 * HP];   // h_t, bf16, padded rows
  __shared__ float gat [16 * GP];   // pre-activation gates [16 x 256]
  __shared__ float cst [16 * HC];   // cell state, f32

  // ---- Preload weight (B-matrix) fragments into registers, once. ----------
  // B[k,n] = W[n_glob, k]; lane holds N = n_glob, elements e -> K = k0+e (+8 for e>=8),
  // k0 = kk*32 + hi*8 (table order, matching the A fragments).
  v16bf Bhh[2][2], Bih[2][2];
  float bias[2];
  for (int nt = 0; nt < 2; ++nt) {
    const int n = (wave * 2 + nt) * 16 + row;   // global gate column
    bias[nt] = bih[n] + bhh[n];
    for (int kk = 0; kk < 2; ++kk) {
      const int k0 = kk * 32 + hi * 8;
      v16bf bh, bi;
      #pragma unroll
      for (int e = 0; e < 16; ++e) {
        const int k = k0 + e + (e >= 8 ? 8 : 0);
        bh[e] = (bf16)Whh[n * H_ + k];
        bi[e] = (k < Din) ? (bf16)Wih[n * Din + k] : (bf16)0.0f;
      }
      Bhh[nt][kk] = bh;
      Bih[nt][kk] = bi;
    }
  }

  for (int i = tid; i < 16 * HC; i += 256) cst[i]  = 0.0f;
  for (int i = tid; i < 16 * HP; i += 256) hlds[i] = (bf16)0.0f;
  __syncthreads();

  unsigned* const srcflag = (layer > 0) ? &flags[(layer - 1) * 16 + btile] : nullptr;
  unsigned* const dstflag = (layer < 2) ? &flags[layer * 16 + btile] : nullptr;
  const size_t rowidx = (size_t)(btile * 16 + row) * T_;

  #pragma unroll 1
  for (int t = 0; t < T_; ++t) {
    // ---- wait for producer layer's timestep t (pipeline handshake) --------
    if (layer > 0) {
      if (tid == 0)
        while (__hip_atomic_load(srcflag, __ATOMIC_ACQUIRE, __HIP_MEMORY_SCOPE_AGENT) <= (unsigned)t)
          __builtin_amdgcn_s_sleep(2);
      __syncthreads();
    }

    // ---- gates = bias + h_{t-1} @ Whh^T + x_t @ Wih^T ---------------------
    v8f acc0, acc1;
    #pragma unroll
    for (int r = 0; r < 8; ++r) { acc0[r] = bias[0]; acc1[r] = bias[1]; }

    #pragma unroll
    for (int kk = 0; kk < 2; ++kk) {         // recurrent GEMM, K = 64
      const int base = kk * 32 + hi * 8;
      AU a;
      a.h8[0] = *(const v8bf*)&hlds[row * HP + base];
      a.h8[1] = *(const v8bf*)&hlds[row * HP + base + 16];
      acc0 = WMMA_BF16(a.v, Bhh[0][kk], acc0);
      acc1 = WMMA_BF16(a.v, Bhh[1][kk], acc1);
    }

    if (layer == 0) {                         // input GEMM, K = 18 (padded to 32)
      const float* xr = x + (rowidx + t) * D_;
      v16bf a;
      #pragma unroll
      for (int e = 0; e < 16; ++e) {
        const int d = e + (e >= 8 ? 8 : 0) + hi * 8;
        a[e] = (d < D_) ? (bf16)xr[d] : (bf16)0.0f;
      }
      acc0 = WMMA_BF16(a, Bih[0][0], acc0);
      acc1 = WMMA_BF16(a, Bih[1][0], acc1);
    } else {                                  // input GEMM, K = 64 (prev layer h)
      const bf16* ir = buf + (rowidx + t) * H_;
      #pragma unroll
      for (int kk = 0; kk < 2; ++kk) {
        const int base = kk * 32 + hi * 8;
        AU a;
        a.h8[0] = *(const v8bf*)&ir[base];
        a.h8[1] = *(const v8bf*)&ir[base + 16];
        acc0 = WMMA_BF16(a.v, Bih[0][kk], acc0);
        acc1 = WMMA_BF16(a.v, Bih[1][kk], acc1);
      }
    }

    // ---- scatter C fragments (M = r + hi*8, N = lane&15 + 16*ntile) -------
    {
      const int n0 = (wave * 2 + 0) * 16 + row;
      const int n1 = (wave * 2 + 1) * 16 + row;
      #pragma unroll
      for (int r = 0; r < 8; ++r) {
        gat[(r + hi * 8) * GP + n0] = acc0[r];
        gat[(r + hi * 8) * GP + n1] = acc1[r];
      }
    }
    __syncthreads();

    // ---- elementwise cell update: 1024 cells / 256 threads = 4 each -------
    {
      const int m  = tid >> 4;
      const int j0 = (tid & 15) * 4;
      bf16* orow = buf + ((size_t)(btile * 16 + m) * T_ + t) * H_;
      #pragma unroll
      for (int jj = 0; jj < 4; ++jj) {
        const int j = j0 + jj;
        const float gi = fsig (gat[m * GP + 0 * H_ + j]);
        const float gf = fsig (gat[m * GP + 1 * H_ + j]);
        const float gg = ftanh(gat[m * GP + 2 * H_ + j]);
        const float go = fsig (gat[m * GP + 3 * H_ + j]);
        const float c  = gf * cst[m * HC + j] + gi * gg;
        cst[m * HC + j] = c;
        const float h  = go * ftanh(c);
        hlds[m * HP + j] = (bf16)h;
        orow[j] = (bf16)h;                 // layer output for consumer / head
      }
    }
    __threadfence();       // each thread: stores visible at agent scope
    __syncthreads();
    if (dstflag && tid == 0)
      __hip_atomic_store(dstflag, (unsigned)(t + 1), __ATOMIC_RELEASE, __HIP_MEMORY_SCOPE_AGENT);
  }
}

// ---- Linear head: out[b,t] = dot(h3[b,t,:], W_out) + b_out -----------------
__global__ __launch_bounds__(256)
void lstm3_head(const bf16* __restrict__ buf, const float* __restrict__ W_out,
                const float* __restrict__ b_out, float* __restrict__ out)
{
  const int idx = blockIdx.x * 256 + threadIdx.x;   // over B*T
  if (idx >= B_ * T_) return;
  const bf16* r = buf + (size_t)idx * H_;
  float acc = 0.0f;
  #pragma unroll
  for (int k = 0; k < H_; ++k) acc += (float)r[k] * W_out[k];
  out[idx] = acc + b_out[0];
}

extern "C" void kernel_launch(void* const* d_in, const int* in_sizes, int n_in,
                              void* d_out, int out_size, void* d_ws, size_t ws_size,
                              hipStream_t stream) {
  (void)in_sizes; (void)n_in; (void)out_size; (void)ws_size;
  const float* x      = (const float*)d_in[0];
  const float* W_ih0  = (const float*)d_in[1];
  const float* W_hh0  = (const float*)d_in[2];
  const float* b_ih0  = (const float*)d_in[3];
  const float* b_hh0  = (const float*)d_in[4];
  const float* W_ih12 = (const float*)d_in[5];
  const float* W_hh12 = (const float*)d_in[6];
  const float* b_ih12 = (const float*)d_in[7];
  const float* b_hh12 = (const float*)d_in[8];
  const float* W_out  = (const float*)d_in[9];
  const float* b_out  = (const float*)d_in[10];

  bf16* buf = (bf16*)d_ws;                                   // [B,T,H] bf16, 32 MB
  const size_t flags_off = (size_t)B_ * T_ * H_ * sizeof(bf16);
  unsigned* flags = (unsigned*)((char*)d_ws + flags_off);    // 32 handshake flags

  hipMemsetAsync(flags, 0, 64 * sizeof(unsigned), stream);   // reset pipeline state

  lstm3_fused<<<dim3(48), dim3(256), 0, stream>>>(
      x, W_ih0, W_hh0, b_ih0, b_hh0, W_ih12, W_hh12, b_ih12, b_hh12, buf, flags);

  lstm3_head<<<dim3((B_ * T_) / 256), dim3(256), 0, stream>>>(
      buf, W_out, b_out, (float*)d_out);
}
